// GRU_45028437131358
// MI455X (gfx1250) — matrix operand order repaired and checked by reading, not compiled, not run on hardware
//
#include <hip/hip_runtime.h>

typedef __bf16 bf16_t;
typedef __bf16 v16bf __attribute__((ext_vector_type(16)));
typedef __bf16 v8bf  __attribute__((ext_vector_type(8)));
typedef float  v8f   __attribute__((ext_vector_type(8)));

#define HSZ 512
#define BSZ 256
#define TSZ 512

#define L0_BUF 20480   // A tile 64*80 + W tile 3*64*80
#define L1_BUF 40960   // 2 A tiles + 2 W tiles

__device__ __forceinline__ unsigned short f2bf(float f) {
    unsigned u = __float_as_uint(f);
    u += 0x7FFFu + ((u >> 16) & 1u);   // round-to-nearest-even
    return (unsigned short)(u >> 16);
}

__device__ __forceinline__ float sigmoidf_(float v) {
    return 1.0f / (1.0f + __expf(-v));
}

// one 16B async copy global -> LDS per lane (ASYNCcnt-tracked, no VGPR data)
__device__ __forceinline__ void async_cp16(unsigned lds_off, const void* g) {
    asm volatile("global_load_async_to_lds_b128 %0, %1, off"
                 :: "v"(lds_off), "v"((unsigned long long)(size_t)g)
                 : "memory");
}

union frag_t { v16bf v; v8bf h[2]; };

__device__ __forceinline__ void ld_frag(frag_t& f, const char* p, int gap) {
    f.h[0] = *(const v8bf*)(p);
    f.h[1] = *(const v8bf*)(p + gap);
}

// ---- fp32 -> bf16 weight conversion (runs once per call, weights then live in L2) ----
__global__ void k_f32_to_bf16(const float* __restrict__ src,
                              unsigned short* __restrict__ dst, int n) {
    int i = blockIdx.x * blockDim.x + threadIdx.x;
    if (i < n) dst[i] = f2bf(src[i]);
}

// ---- GRU layer 0: gh = h0 @ Wh0^T via bf16 WMMA, async-LDS double-buffered ----
__global__ __launch_bounds__(256)
void k_gru_layer0(const float* __restrict__ x, int t,
                  const float* __restrict__ Wi0, const float* __restrict__ bi0,
                  const float* __restrict__ bh0,
                  const bf16_t* __restrict__ wh0,      // [3][H][H] bf16
                  const float* __restrict__ h_cur,     // [B][H] fp32 master
                  const bf16_t* __restrict__ h_cur_bf, // [B][H] bf16 shadow
                  float* __restrict__ h_nxt,
                  unsigned short* __restrict__ h_nxt_bf)
{
    extern __shared__ char smem[];
    const int tid    = threadIdx.x;
    const int lane   = tid & 31;
    const int wave   = tid >> 5;
    const int mstrip = wave & 3;          // 4 M strips of 16
    const int nhalf  = wave >> 2;         // 2 N halves of 32
    const int mblk = blockIdx.y * 64;
    const int nblk = blockIdx.x * 64;
    const int m0   = mblk + mstrip * 16;
    const int n0   = nblk + nhalf * 32;
    const int lhi  = lane >> 4;
    const int llo  = lane & 15;
    const unsigned lds0 = (unsigned)(size_t)(void*)smem;

    const int srow = tid >> 2;            // staging: 64 rows x 4 chunks of 16B
    const int sch  = tid & 3;

    auto stage = [&](int kc, int b) {
        const unsigned base = lds0 + b * L0_BUF;
        const int go = kc * 32 + sch * 8;
        async_cp16(base + srow * 80 + sch * 16,
                   h_cur_bf + (mblk + srow) * HSZ + go);
        #pragma unroll
        for (int g = 0; g < 3; ++g)
            async_cp16(base + 5120 + (g * 64 + srow) * 80 + sch * 16,
                       wh0 + (size_t)g * HSZ * HSZ + (nblk + srow) * HSZ + go);
    };

    v8f acc[3][2];
    #pragma unroll
    for (int g = 0; g < 3; ++g)
        #pragma unroll
        for (int j = 0; j < 2; ++j)
            #pragma unroll
            for (int e = 0; e < 8; ++e) acc[g][j][e] = 0.0f;

    stage(0, 0);
    const int NK = HSZ / 32;
    for (int kt = 0; kt < NK; ++kt) {
        const int b = kt & 1;
        if (kt + 1 < NK) {
            stage(kt + 1, b ^ 1);
            asm volatile("s_wait_asynccnt 0x4" ::: "memory");  // keep prefetch in flight
        } else {
            asm volatile("s_wait_asynccnt 0x0" ::: "memory");
        }
        __syncthreads();

        const char* buf = smem + b * L0_BUF;
        frag_t afrag;
        ld_frag(afrag, buf + (mstrip * 16 + llo) * 80 + lhi * 16, 32);

        // batch ALL 6 B-fragment loads into distinct regs, then the WMMA chain
        frag_t bfrag[3][2];
        #pragma unroll
        for (int g = 0; g < 3; ++g)
            #pragma unroll
            for (int j = 0; j < 2; ++j)
                ld_frag(bfrag[g][j],
                        buf + 5120 + (g * 64 + nhalf * 32 + j * 16 + llo) * 80 + lhi * 16, 32);
        #pragma unroll
        for (int g = 0; g < 3; ++g)
            #pragma unroll
            for (int j = 0; j < 2; ++j)
                acc[g][j] = __builtin_amdgcn_wmma_f32_16x16x32_bf16(
                    false, afrag.v, false, bfrag[g][j].v, (short)0, acc[g][j], false, false);
        __syncthreads();
    }

    // pointwise GRU combine (fp32); C layout: VGPR r -> M = r + 8*lhi, N = llo
    #pragma unroll
    for (int j = 0; j < 2; ++j) {
        const int n = n0 + j * 16 + llo;
        const float bhr = bh0[0 * HSZ + n];
        const float bhz = bh0[1 * HSZ + n];
        const float bhn = bh0[2 * HSZ + n];
        const float wir = Wi0[0 * HSZ + n], bir  = bi0[0 * HSZ + n];
        const float wiz = Wi0[1 * HSZ + n], biz  = bi0[1 * HSZ + n];
        const float win = Wi0[2 * HSZ + n], bin_ = bi0[2 * HSZ + n];
        #pragma unroll
        for (int r = 0; r < 8; ++r) {
            const int m  = m0 + r + lhi * 8;
            const float xb  = x[m * TSZ + t];
            const float ghr = acc[0][j][r] + bhr;
            const float ghz = acc[1][j][r] + bhz;
            const float ghn = acc[2][j][r] + bhn;
            const float rg  = sigmoidf_(wir * xb + bir + ghr);
            const float zg  = sigmoidf_(wiz * xb + biz + ghz);
            const float ng  = tanhf(win * xb + bin_ + rg * ghn);
            const float hp  = h_cur[m * HSZ + n];
            const float hn  = (1.0f - zg) * ng + zg * hp;
            h_nxt[m * HSZ + n]    = hn;
            h_nxt_bf[m * HSZ + n] = f2bf(hn);
        }
    }
}

// ---- GRU layer 1: gi = h0_new @ Wi1^T, gh = h1 @ Wh1^T, async-LDS double-buffered ----
__global__ __launch_bounds__(256)
void k_gru_layer1(const float* __restrict__ bi1, const float* __restrict__ bh1,
                  const bf16_t* __restrict__ wi1,      // [3][H][H]
                  const bf16_t* __restrict__ wh1,      // [3][H][H]
                  const bf16_t* __restrict__ in_bf,    // layer-0 new h, bf16
                  const float* __restrict__ h_cur, const bf16_t* __restrict__ h_cur_bf,
                  float* __restrict__ h_nxt, unsigned short* __restrict__ h_nxt_bf)
{
    extern __shared__ char smem[];
    const int tid    = threadIdx.x;
    const int lane   = tid & 31;
    const int wave   = tid >> 5;
    const int mstrip = wave & 3;
    const int nhalf  = wave >> 2;
    const int mblk = blockIdx.y * 64;
    const int nblk = blockIdx.x * 64;
    const int m0   = mblk + mstrip * 16;
    const int n0   = nblk + nhalf * 32;
    const int lhi  = lane >> 4;
    const int llo  = lane & 15;
    const unsigned lds0 = (unsigned)(size_t)(void*)smem;

    const int srow = tid >> 2;
    const int sch  = tid & 3;

    auto stage = [&](int kc, int b) {
        const unsigned base = lds0 + b * L1_BUF;
        const int go = kc * 32 + sch * 8;
        async_cp16(base +        srow * 80 + sch * 16, in_bf    + (mblk + srow) * HSZ + go);
        async_cp16(base + 5120 + srow * 80 + sch * 16, h_cur_bf + (mblk + srow) * HSZ + go);
        #pragma unroll
        for (int g = 0; g < 3; ++g) {
            async_cp16(base + 10240 + (g * 64 + srow) * 80 + sch * 16,
                       wi1 + (size_t)g * HSZ * HSZ + (nblk + srow) * HSZ + go);
            async_cp16(base + 25600 + (g * 64 + srow) * 80 + sch * 16,
                       wh1 + (size_t)g * HSZ * HSZ + (nblk + srow) * HSZ + go);
        }
    };

    v8f acci[3][2], acch[3][2];
    #pragma unroll
    for (int g = 0; g < 3; ++g)
        #pragma unroll
        for (int j = 0; j < 2; ++j)
            #pragma unroll
            for (int e = 0; e < 8; ++e) { acci[g][j][e] = 0.0f; acch[g][j][e] = 0.0f; }

    stage(0, 0);
    const int NK = HSZ / 32;
    for (int kt = 0; kt < NK; ++kt) {
        const int b = kt & 1;
        if (kt + 1 < NK) {
            stage(kt + 1, b ^ 1);
            asm volatile("s_wait_asynccnt 0x8" ::: "memory");
        } else {
            asm volatile("s_wait_asynccnt 0x0" ::: "memory");
        }
        __syncthreads();

        const char* buf = smem + b * L1_BUF;
        frag_t afin, afh;
        ld_frag(afin, buf +        (mstrip * 16 + llo) * 80 + lhi * 16, 32);
        ld_frag(afh,  buf + 5120 + (mstrip * 16 + llo) * 80 + lhi * 16, 32);

        // per gate: batch 4 fragment loads (wi/wh x 2 N-tiles), then 4 WMMAs
        #pragma unroll
        for (int g = 0; g < 3; ++g) {
            frag_t bfi[2], bfh[2];
            #pragma unroll
            for (int j = 0; j < 2; ++j) {
                const int colp = (g * 64 + nhalf * 32 + j * 16 + llo) * 80 + lhi * 16;
                ld_frag(bfi[j], buf + 10240 + colp, 32);
                ld_frag(bfh[j], buf + 25600 + colp, 32);
            }
            #pragma unroll
            for (int j = 0; j < 2; ++j) {
                acci[g][j] = __builtin_amdgcn_wmma_f32_16x16x32_bf16(
                    false, afin.v, false, bfi[j].v, (short)0, acci[g][j], false, false);
                acch[g][j] = __builtin_amdgcn_wmma_f32_16x16x32_bf16(
                    false, afh.v, false, bfh[j].v, (short)0, acch[g][j], false, false);
            }
        }
        __syncthreads();
    }

    #pragma unroll
    for (int j = 0; j < 2; ++j) {
        const int n = n0 + j * 16 + llo;
        const float bir  = bi1[0 * HSZ + n], bhr = bh1[0 * HSZ + n];
        const float biz  = bi1[1 * HSZ + n], bhz = bh1[1 * HSZ + n];
        const float bin_ = bi1[2 * HSZ + n], bhn = bh1[2 * HSZ + n];
        #pragma unroll
        for (int r = 0; r < 8; ++r) {
            const int m  = m0 + r + lhi * 8;
            const float gir = acci[0][j][r] + bir;
            const float giz = acci[1][j][r] + biz;
            const float gin = acci[2][j][r] + bin_;
            const float ghr = acch[0][j][r] + bhr;
            const float ghz = acch[1][j][r] + bhz;
            const float ghn = acch[2][j][r] + bhn;
            const float rg  = sigmoidf_(gir + ghr);
            const float zg  = sigmoidf_(giz + ghz);
            const float ng  = tanhf(gin + rg * ghn);
            const float hp  = h_cur[m * HSZ + n];
            const float hn  = (1.0f - zg) * ng + zg * hp;
            h_nxt[m * HSZ + n]    = hn;
            h_nxt_bf[m * HSZ + n] = f2bf(hn);
        }
    }
}

// ---- final FC: (256 x 512) @ (512 x 2) + bias, tiny -> plain VALU ----
__global__ void k_fc(const float* __restrict__ h, const float* __restrict__ fc_w,
                     const float* __restrict__ fc_b, float* __restrict__ out) {
    int i = blockIdx.x * blockDim.x + threadIdx.x;
    if (i >= BSZ * 2) return;
    int b = i >> 1, o = i & 1;
    float s = fc_b[o];
    for (int k = 0; k < HSZ; ++k) s += h[b * HSZ + k] * fc_w[o * HSZ + k];
    out[i] = s;
}

extern "C" void kernel_launch(void* const* d_in, const int* in_sizes, int n_in,
                              void* d_out, int out_size, void* d_ws, size_t ws_size,
                              hipStream_t stream) {
    const float* x       = (const float*)d_in[0];  // (B,T)
    const float* Wi0     = (const float*)d_in[1];  // (3,H,1)
    const float* bi0     = (const float*)d_in[2];  // (3,H)
    const float* Wi_rest = (const float*)d_in[3];  // (1,3,H,H)
    const float* bi_rest = (const float*)d_in[4];  // (1,3,H)
    const float* Wh      = (const float*)d_in[5];  // (2,3,H,H)
    const float* bh      = (const float*)d_in[6];  // (2,3,H)
    const float* fc_w    = (const float*)d_in[7];  // (2,H)
    const float* fc_b    = (const float*)d_in[8];  // (2,)

    char* ws = (char*)d_ws;
    unsigned short* wh_bf  = (unsigned short*)(ws);             // 2*3*H*H bf16 = 3 MB
    unsigned short* wi1_bf = (unsigned short*)(ws + 3145728);   // 3*H*H bf16 = 1.5 MB
    float*          h0f    = (float*)(ws + 4718592);            // 2 ping-pong buffers
    float*          h1f    = (float*)(ws + 5767168);
    unsigned short* h0b    = (unsigned short*)(ws + 6815744);
    unsigned short* h1b    = (unsigned short*)(ws + 7340032);

    const int nWh = 2 * 3 * HSZ * HSZ;
    const int nWi = 3 * HSZ * HSZ;
    k_f32_to_bf16<<<(nWh + 255) / 256, 256, 0, stream>>>(Wh, wh_bf, nWh);
    k_f32_to_bf16<<<(nWi + 255) / 256, 256, 0, stream>>>(Wi_rest, wi1_bf, nWi);

    // zero the t=0 (index 0) hidden-state buffers
    hipMemsetAsync(h0f, 0, BSZ * HSZ * sizeof(float), stream);
    hipMemsetAsync(h1f, 0, BSZ * HSZ * sizeof(float), stream);
    hipMemsetAsync(h0b, 0, BSZ * HSZ * sizeof(unsigned short), stream);
    hipMemsetAsync(h1b, 0, BSZ * HSZ * sizeof(unsigned short), stream);

    dim3 grid(HSZ / 64, BSZ / 64);   // (8, 4), 256 threads = 8 wave32 each
    const int HB = BSZ * HSZ;
    for (int t = 0; t < TSZ; ++t) {
        int cur = t & 1, nxt = cur ^ 1;
        k_gru_layer0<<<grid, 256, 2 * L0_BUF, stream>>>(
            x, t, Wi0, bi0, bh /* layer 0 */,
            (const bf16_t*)wh_bf,
            h0f + cur * HB, (const bf16_t*)(h0b + cur * HB),
            h0f + nxt * HB, h0b + nxt * HB);
        k_gru_layer1<<<grid, 256, 2 * L1_BUF, stream>>>(
            bi_rest, bh + 3 * HSZ /* layer 1 */,
            (const bf16_t*)wi1_bf, (const bf16_t*)(wh_bf + 3 * HSZ * HSZ),
            (const bf16_t*)(h0b + nxt * HB),
            h1f + cur * HB, (const bf16_t*)(h1b + cur * HB),
            h1f + nxt * HB, h1b + nxt * HB);
    }
    // after t = 511 (cur=1), final state lives in buffer 0
    k_fc<<<2, 256, 0, stream>>>(h1f, fc_w, fc_b, (float*)d_out);
}
